// MainModel_19069654794280
// MI455X (gfx1250) — matrix operand order repaired and checked by reading, not compiled, not run on hardware
//
#include <hip/hip_runtime.h>

// ---------------------------------------------------------------------------
// Problem constants (match reference)
// ---------------------------------------------------------------------------
#define BQ    32        // batch
#define DIM   512       // embedding dim
#define NIMG  50000
#define NKN   8192
#define NCLS  32        // C
#define RTOP  16        // image retrieval
#define KTOP  8         // knowledge retrieval
#define NEGV  (-1e9f)

typedef unsigned short u16;
typedef __attribute__((ext_vector_type(16))) __bf16 v16bf;
typedef __attribute__((ext_vector_type(8)))  __bf16 v8bf;
typedef __attribute__((ext_vector_type(8)))  float  v8f;

// round-to-nearest-even fp32 -> bf16 (bit pattern)
static __device__ __forceinline__ u16 f2bf(float f) {
    unsigned int u = __float_as_uint(f);
    u = (u + 0x7FFFu + ((u >> 16) & 1u)) >> 16;
    return (u16)u;
}

// Build WMMA A-operand fragment for this lane: 16 contiguous bf16 at p[0..15].
// (contiguous K order: lane<16 holds K={0..15}; lane+16 holds
//  K={16..31} — caller applies the half-wave +16 element base offset.)
template <typename PT>
static __device__ __forceinline__ v16bf load_afrag(const PT* p) {
    v8bf lo = *(const v8bf*)(p);
    v8bf hi = *(const v8bf*)(p + 8);
    v16bf r;
#pragma unroll
    for (int i = 0; i < 8; ++i) { r[i] = lo[i]; r[i + 8] = hi[i]; }
    return r;
}

// ---------------------------------------------------------------------------
// K1: fp32 -> bf16 conversion
// ---------------------------------------------------------------------------
__global__ void k_f2bf(const float* __restrict__ src, u16* __restrict__ dst, int n) {
    int i = blockIdx.x * blockDim.x + threadIdx.x;
    if (i < n) dst[i] = f2bf(src[i]);
}

// ---------------------------------------------------------------------------
// K2: sims[BQ][NIMG] = Qbf16 (32x512) @ Ibf16^T (512xNIMG) via WMMA bf16.
// One wave per 16-image tile; two accumulators cover M=32.
// ---------------------------------------------------------------------------
__global__ __launch_bounds__(32) void k_sims_wmma(const u16* __restrict__ qb,
                                                  const u16* __restrict__ imgb,
                                                  float* __restrict__ sims) {
    const int lane = threadIdx.x;
    const int half = lane >> 4;    // 0|1
    const int l16  = lane & 15;
    const int n0   = blockIdx.x * 16;

    const u16* a0p  = qb + (size_t)l16 * DIM + half * 16;         // queries rows 0..15
    const u16* a1p  = qb + (size_t)(l16 + 16) * DIM + half * 16;  // queries rows 16..31
    const u16* brow = imgb + (size_t)(n0 + l16) * DIM + half * 16; // image rows (B operand)

    v8f c0 = {}; v8f c1 = {};
#pragma unroll
    for (int k0 = 0; k0 < DIM; k0 += 32) {
        v16bf a0 = load_afrag(a0p + k0);
        v16bf a1 = load_afrag(a1p + k0);
        v16bf bb = *(const v16bf*)(brow + k0);
        c0 = __builtin_amdgcn_wmma_f32_16x16x32_bf16(false, a0, false, bb, (short)0, c0, false, false);
        c1 = __builtin_amdgcn_wmma_f32_16x16x32_bf16(false, a1, false, bb, (short)0, c1, false, false);
    }

    // D layout: VGPR j -> M = j + 8*half, N = l16
    const int n  = n0 + l16;
    const int mb = half * 8;
#pragma unroll
    for (int j = 0; j < 8; ++j) {
        sims[(size_t)(mb + j) * NIMG + n]      = c0[j];
        sims[(size_t)(mb + j + 16) * NIMG + n] = c1[j];
    }
}

// ---------------------------------------------------------------------------
// K3: per-(b,c) masked top-16 over sims, softmax, fused image + output.
// ---------------------------------------------------------------------------
__global__ __launch_bounds__(256) void k_img_topk(const float* __restrict__ sims,
                                                  const int*  __restrict__ labels,
                                                  const float* __restrict__ imgf,
                                                  const float* __restrict__ qf,
                                                  int*  __restrict__ oidx,
                                                  float* __restrict__ ow,
                                                  float* __restrict__ out_img) {
    __shared__ float sv[256 * 16];
    __shared__ int   si[256 * 16];
    __shared__ float lw[RTOP];
    __shared__ int   li[RTOP];

    const int tid = threadIdx.x;
    const int b = blockIdx.x / NCLS, c = blockIdx.x % NCLS;
    const float* srow = sims + (size_t)b * NIMG;

    float tv[16]; int ti[16];
#pragma unroll
    for (int i = 0; i < 16; ++i) { tv[i] = NEGV; ti[i] = 0; }

    for (int n = tid; n < NIMG; n += 256) {
        if (labels[n] == c) {
            float v = srow[n];
            if (v > tv[15]) {
                int p = 15;
                while (p > 0 && tv[p - 1] < v) { tv[p] = tv[p - 1]; ti[p] = ti[p - 1]; --p; }
                tv[p] = v; ti[p] = n;
            }
        }
    }
#pragma unroll
    for (int i = 0; i < 16; ++i) { sv[tid * 16 + i] = tv[i]; si[tid * 16 + i] = ti[i]; }

    // tree-merge 256 sorted(desc) lists down to one
    for (int stride = 128; stride >= 1; stride >>= 1) {
        __syncthreads();
        if (tid < stride) {
            const int a = tid * 16, bb = (tid + stride) * 16;
            float ovv[16]; int oii[16]; int pa = 0, pb = 0;
#pragma unroll
            for (int o = 0; o < 16; ++o) {
                float va = sv[a + pa], vb = sv[bb + pb];
                if (va >= vb) { ovv[o] = va; oii[o] = si[a + pa]; ++pa; }
                else          { ovv[o] = vb; oii[o] = si[bb + pb]; ++pb; }
            }
#pragma unroll
            for (int o = 0; o < 16; ++o) { sv[a + o] = ovv[o]; si[a + o] = oii[o]; }
        }
    }
    __syncthreads();

    if (tid < RTOP) {
        float m = sv[0];                       // sorted desc -> max first
        float s = 0.f;
#pragma unroll
        for (int j = 0; j < RTOP; ++j) s += __expf(sv[j] - m);
        float w = __expf(sv[tid] - m) / s;
        int gi = si[tid];
        oidx[(size_t)(b * NCLS + c) * RTOP + tid] = gi;
        ow  [(size_t)(b * NCLS + c) * RTOP + tid] = w;
        lw[tid] = w; li[tid] = gi;
    }
    __syncthreads();

    for (int d = tid; d < DIM; d += 256) {
        float acc = 0.f;
#pragma unroll
        for (int r = 0; r < RTOP; ++r) acc += lw[r] * imgf[(size_t)li[r] * DIM + d];
        out_img[(size_t)(b * NCLS + c) * DIM + d] = 0.5f * (qf[(size_t)b * DIM + d] + acc);
    }
}

// ---------------------------------------------------------------------------
// K4: deterministic per-class compaction of knowledge indices.
// ---------------------------------------------------------------------------
__global__ void k_kn_compact(const int* __restrict__ klabels,
                             int* __restrict__ list, int* __restrict__ cnt) {
    int c = blockIdx.x;
    if (threadIdx.x == 0) {
        int m = 0;
        for (int n = 0; n < NKN; ++n)
            if (klabels[n] == c) list[c * NKN + m++] = n;
        cnt[c] = m;
    }
}

// ---------------------------------------------------------------------------
// K5: per-(b,c): WMMA scores of 16 retrieved images vs class-c knowledge,
// per-r top-8 + softmax, double-weighted fusion -> enhanced knowledge.
// ---------------------------------------------------------------------------
__global__ __launch_bounds__(32) void k_kn_retrieve(const u16* __restrict__ imgb,
                                                    const u16* __restrict__ knb,
                                                    const float* __restrict__ knf,
                                                    const float* __restrict__ qf,
                                                    const int*  __restrict__ oidx,
                                                    const float* __restrict__ ow,
                                                    const int*  __restrict__ list,
                                                    const int*  __restrict__ cntp,
                                                    float* __restrict__ out_kn) {
    __shared__ u16   sA[RTOP * DIM];       // 16 KB: retrieved image rows (bf16)
    __shared__ float sc[16][16];           // score tile
    __shared__ float s_iw[RTOP];
    __shared__ int   s_ii[RTOP];
    __shared__ float skw[RTOP][KTOP];
    __shared__ int   skid[RTOP][KTOP];

    const int lane = threadIdx.x;
    const int b = blockIdx.x / NCLS, c = blockIdx.x % NCLS;
    const int half = lane >> 4, l16 = lane & 15;

    if (lane < RTOP) {
        s_ii[lane] = oidx[(size_t)(b * NCLS + c) * RTOP + lane];
        s_iw[lane] = ow  [(size_t)(b * NCLS + c) * RTOP + lane];
    }
    __syncthreads();

    // stage A (16 gathered image rows) into LDS, 16B chunks
    for (int i = lane; i < RTOP * (DIM / 8); i += 32) {
        int row = i / (DIM / 8), col = (i % (DIM / 8)) * 8;
        *(v8bf*)&sA[row * DIM + col] = *(const v8bf*)(imgb + (size_t)s_ii[row] * DIM + col);
    }
    __syncthreads();

    const int  cnt = cntp[c];
    const int* lst = list + c * NKN;

    float kv[KTOP]; int ki[KTOP];
#pragma unroll
    for (int i = 0; i < KTOP; ++i) { kv[i] = NEGV; ki[i] = 0; }

    const u16* arow = &sA[l16 * DIM + half * 16];
    const int ntiles = (cnt + 15) / 16;
    for (int t = 0; t < ntiles; ++t) {
        int jglob = t * 16 + l16;
        int rowid = lst[jglob < cnt ? jglob : (cnt - 1)];
        const u16* brow = knb + (size_t)rowid * DIM + half * 16;

        v8f acc = {};
#pragma unroll
        for (int k0 = 0; k0 < DIM; k0 += 32) {
            v16bf a  = load_afrag(arow + k0);
            v16bf bb = *(const v16bf*)(brow + k0);
            acc = __builtin_amdgcn_wmma_f32_16x16x32_bf16(false, a, false, bb, (short)0, acc, false, false);
        }
        const int mb = half * 8;
#pragma unroll
        for (int j = 0; j < 8; ++j) sc[mb + j][l16] = acc[j];
        __syncthreads();

        if (lane < RTOP) {
#pragma unroll
            for (int j = 0; j < 16; ++j) {
                int g = t * 16 + j;
                if (g < cnt) {
                    float v = sc[lane][j];
                    if (v > kv[KTOP - 1]) {
                        int p = KTOP - 1;
                        while (p > 0 && kv[p - 1] < v) { kv[p] = kv[p - 1]; ki[p] = ki[p - 1]; --p; }
                        kv[p] = v; ki[p] = lst[g];
                    }
                }
            }
        }
        __syncthreads();
    }

    if (lane < RTOP) {
        float m = kv[0];
        float s = 0.f;
#pragma unroll
        for (int j = 0; j < KTOP; ++j) s += __expf(kv[j] - m);
#pragma unroll
        for (int j = 0; j < KTOP; ++j) { skw[lane][j] = __expf(kv[j] - m) / s; skid[lane][j] = ki[j]; }
    }
    __syncthreads();

    for (int d = lane; d < DIM; d += 32) {
        float acc = 0.f;
#pragma unroll
        for (int r = 0; r < RTOP; ++r) {
            float inner = 0.f;
#pragma unroll
            for (int k = 0; k < KTOP; ++k)
                inner += skw[r][k] * knf[(size_t)skid[r][k] * DIM + d];
            acc += s_iw[r] * inner;
        }
        out_kn[(size_t)(b * NCLS + c) * DIM + d] = 0.5f * (qf[(size_t)b * DIM + d] + acc);
    }
}

// ---------------------------------------------------------------------------
// Host-side launcher
// ---------------------------------------------------------------------------
extern "C" void kernel_launch(void* const* d_in, const int* in_sizes, int n_in,
                              void* d_out, int out_size, void* d_ws, size_t ws_size,
                              hipStream_t stream) {
    const float* qf   = (const float*)d_in[0];   // [32,512]
    const float* imgf = (const float*)d_in[1];   // [50000,512]
    const float* knf  = (const float*)d_in[2];   // [8192,512]
    const int*   ilab = (const int*)  d_in[3];   // [50000]
    const int*   klab = (const int*)  d_in[4];   // [8192]

    float* out_img = (float*)d_out;                                  // [B,C,D]
    float* out_kn  = (float*)d_out + (size_t)BQ * NCLS * DIM;        // [B,C,D]

    // workspace carve-out (256B aligned); total ~67 MB
    char* base = (char*)d_ws;
    size_t off = 0;
    auto carve = [&](size_t bytes) -> char* {
        char* p = base + off;
        off = (off + bytes + 255) & ~(size_t)255;
        return p;
    };
    u16*   qb      = (u16*)  carve((size_t)BQ * DIM * sizeof(u16));
    u16*   imgb    = (u16*)  carve((size_t)NIMG * DIM * sizeof(u16));
    u16*   knb     = (u16*)  carve((size_t)NKN * DIM * sizeof(u16));
    float* sims    = (float*)carve((size_t)BQ * NIMG * sizeof(float));
    int*   img_idx = (int*)  carve((size_t)BQ * NCLS * RTOP * sizeof(int));
    float* img_w   = (float*)carve((size_t)BQ * NCLS * RTOP * sizeof(float));
    int*   kn_cnt  = (int*)  carve((size_t)NCLS * sizeof(int));
    int*   kn_list = (int*)  carve((size_t)NCLS * NKN * sizeof(int));
    (void)ws_size; (void)n_in; (void)in_sizes; (void)out_size;

    // 1) fp32 -> bf16
    {
        int nq = BQ * DIM, ni = NIMG * DIM, nk = NKN * DIM;
        k_f2bf<<<(nq + 255) / 256, 256, 0, stream>>>(qf, qb, nq);
        k_f2bf<<<(ni + 255) / 256, 256, 0, stream>>>(imgf, imgb, ni);
        k_f2bf<<<(nk + 255) / 256, 256, 0, stream>>>(knf, knb, nk);
    }
    // 2) sims = Q @ I^T  (WMMA bf16, 16-image tiles, NIMG = 3125*16)
    k_sims_wmma<<<NIMG / 16, 32, 0, stream>>>(qb, imgb, sims);
    // 3) per-(b,c) image top-16 + softmax + fused/enhanced images
    k_img_topk<<<BQ * NCLS, 256, 0, stream>>>(sims, ilab, imgf, qf, img_idx, img_w, out_img);
    // 4) per-class knowledge compaction
    k_kn_compact<<<NCLS, 32, 0, stream>>>(klab, kn_list, kn_cnt);
    // 5) per-(b,c) knowledge WMMA retrieval + fusion
    k_kn_retrieve<<<BQ * NCLS, 32, 0, stream>>>(imgb, knb, knf, qf, img_idx, img_w,
                                                kn_list, kn_cnt, out_kn);
}
